// CNNMambaFast_31224412242006
// MI455X (gfx1250) — compile-verified, matrix operand order checked by reading, not run on hardware
//
#include <hip/hip_runtime.h>
#include <stdint.h>

// ---------------------------------------------------------------------------
// CNN + Mamba pipeline for MI455X (gfx1250).
// GEMMs run on v_wmma_f32_16x16x32_bf16 (bf16 in, f32 accumulate).
// Each wave accumulates four 16x16 C tiles; A staged 16x64 in LDS per block.
// ---------------------------------------------------------------------------

typedef __attribute__((ext_vector_type(16))) __bf16        bf16x16;
typedef __attribute__((ext_vector_type(8)))  float         f32x8;
typedef __attribute__((ext_vector_type(4)))  unsigned int  u32x4;

union FragAB { bf16x16 v; u32x4 q[2]; };
union FragC  { f32x8   v; float  f[8]; };

#define DEV __device__ __forceinline__

DEV unsigned short f32_to_bf16(float f) {
    unsigned int x = __float_as_uint(f);
    unsigned int r = x + 0x7FFFu + ((x >> 16) & 1u);   // round-to-nearest-even
    return (unsigned short)(r >> 16);
}
DEV float sigm(float x) { return 1.0f / (1.0f + expf(-x)); }

// ---------------------------------------------------------------------------
// Problem constants
// ---------------------------------------------------------------------------
constexpr int BB = 4, TT = 512, CC = 24;
constexpr int MM = BB * TT;            // 2048 rows for all GEMMs
constexpr int DM = 960, DI = 1920;     // d_model, d_inner

// ---------------------------------------------------------------------------
// Workspace layout (bytes)
// ---------------------------------------------------------------------------
constexpr size_t ALN(size_t x) { return (x + 255) & ~size_t(255); }

constexpr size_t O_WB_IN  = 0;                         constexpr size_t S_WB_IN  = (size_t)3*3840*960*2;
constexpr size_t O_WB_X   = ALN(O_WB_IN  + S_WB_IN);   constexpr size_t S_WB_X   = (size_t)3*128*1920*2;
constexpr size_t O_WB_DT  = ALN(O_WB_X   + S_WB_X);    constexpr size_t S_WB_DT  = (size_t)3*1920*64*2;
constexpr size_t O_WB_OUT = ALN(O_WB_DT  + S_WB_DT);   constexpr size_t S_WB_OUT = (size_t)3*960*1920*2;
constexpr size_t O_WB_FC  = ALN(O_WB_OUT + S_WB_OUT);  constexpr size_t S_WB_FC  = (size_t)64*960*2;
constexpr size_t O_SEQF   = ALN(O_WB_FC  + S_WB_FC);   constexpr size_t S_SEQF   = (size_t)MM*DM*4;
constexpr size_t O_SEQB   = ALN(O_SEQF   + S_SEQF);    constexpr size_t S_SEQB   = (size_t)MM*DM*2;
constexpr size_t O_SCR    = ALN(O_SEQB   + S_SEQB);

// Phase B (Mamba) inside SCR
constexpr size_t O_XZ  = O_SCR;                 constexpr size_t S_XZ  = (size_t)MM*3840*4;
constexpr size_t O_XC  = ALN(O_XZ  + S_XZ);     constexpr size_t S_XC  = (size_t)MM*DI*4;
constexpr size_t O_XCB = ALN(O_XC  + S_XC);     constexpr size_t S_XCB = (size_t)MM*DI*2;
constexpr size_t O_DBC = ALN(O_XCB + S_XCB);    constexpr size_t S_DBC = (size_t)MM*128*4;
constexpr size_t O_DTI = ALN(O_DBC + S_DBC);    constexpr size_t S_DTI = (size_t)MM*64*2;
constexpr size_t O_DTP = ALN(O_DTI + S_DTI);    constexpr size_t S_DTP = (size_t)MM*DI*4;
constexpr size_t O_YS  = ALN(O_DTP + S_DTP);    constexpr size_t S_YS  = (size_t)MM*DI*4;
constexpr size_t O_YB  = ALN(O_YS  + S_YS);     constexpr size_t S_YB  = (size_t)MM*DI*2;
constexpr size_t O_OP  = ALN(O_YB  + S_YB);     constexpr size_t S_OP  = (size_t)MM*DM*4;
constexpr size_t O_LOG = ALN(O_OP  + S_OP);     constexpr size_t S_LOG = (size_t)MM*64*4;
constexpr size_t O_END = ALN(O_LOG + S_LOG);

// Phase A (CNN) aliases inside SCR (phase A finishes before phase B starts)
constexpr size_t O_HFLUX = O_SCR;                         // 4*160*512*4   = 1,310,720
constexpr size_t O_R1    = O_SCR + 1310720;               // 4*24*160*512*4 = 31,457,280  (also H1)
constexpr size_t O_R2    = O_R1  + 31457280;              // (also P1)
constexpr size_t O_R3    = O_R2  + 31457280;              // 4*24*80*512*4 = 15,728,640
constexpr size_t O_R4    = O_R3  + 15728640;
constexpr size_t O_H2    = O_SCR;                         // overwrites HFLUX (dead by then)
constexpr size_t O_P2    = O_SCR + 16777216;              // 4*24*40*512*4 = 7,864,320

// ---------------------------------------------------------------------------
// bf16 WMMA GEMM:  C[M,N] = A[M,K] * B[N,K]^T
//   A bf16 row-major (lda), B packed bf16 panel row-major (ldb), C f32 (ldc).
//   K%64==0 (as kTiles%2==0), N%64==0 with zero padding -> no edge guards.
//   Block: 256 thr = 8 waves. Wave w owns N-group (blockIdx.y*8+w) of four
//   16x16 tiles. Per stage: 16x64 A panel -> LDS (one b128/thread), then
//   2 k-steps x 4 N-tiles = 8 WMMAs reusing the staged A fragments.
// ---------------------------------------------------------------------------
__global__ __launch_bounds__(256) void gemm_bf16_wmma(
    const unsigned short* __restrict__ A, const unsigned short* __restrict__ Bw,
    float* __restrict__ C, int lda, int ldb, int ldc, int kTiles, int nTiles)
{
    __shared__ u32x4 ldsA[128];                 // 16 rows x 64 bf16 = 2KB
    const int tid  = threadIdx.x;
    const int wave = tid >> 5;
    const int lane = tid & 31;
    const int half = lane >> 4;                 // ISA A/B K-striping half
    const int mlan = lane & 15;
    const int m0   = blockIdx.x << 4;
    const int grp  = blockIdx.y * 8 + wave;     // group of 4 consecutive N tiles
    const int nGroups = nTiles >> 2;
    // wave-uniform predicate, forced scalar so EXEC stays all-ones around WMMA
    const bool active = __builtin_amdgcn_readfirstlane((int)(grp < nGroups)) != 0;

    FragC acc0, acc1, acc2, acc3;
#pragma unroll
    for (int r = 0; r < 8; ++r) { acc0.f[r] = 0.f; acc1.f[r] = 0.f; acc2.f[r] = 0.f; acc3.f[r] = 0.f; }

    const int gsafe = active ? grp : 0;
    const unsigned short* b0 = Bw + (size_t)(gsafe * 64 + mlan) * ldb;
    const unsigned short* b1 = b0 + (size_t)16 * ldb;
    const unsigned short* b2 = b0 + (size_t)32 * ldb;
    const unsigned short* b3 = b0 + (size_t)48 * ldb;

    const int lrow = tid >> 3, lchunk = tid & 7;        // A loader: 128 threads
    const u32x4* asrc = (const u32x4*)(A + (size_t)(m0 + lrow) * lda);

    const int stages = kTiles >> 1;                     // 64 K per stage
    for (int ks = 0; ks < stages; ++ks) {
        __syncthreads();
        if (tid < 128) ldsA[tid] = asrc[ks * 8 + lchunk];
        __syncthreads();
        if (active) {
            const u32x4* lrowp = (const u32x4*)&ldsA[mlan << 3];
            __builtin_prefetch(b0 + (size_t)(ks + 1) * 64, 0, 1);   // speculative
            __builtin_prefetch(b2 + (size_t)(ks + 1) * 64, 0, 1);
#pragma unroll
            for (int kk = 0; kk < 2; ++kk) {
                FragAB a;
                a.q[0] = lrowp[(kk << 2) + half];
                a.q[1] = lrowp[(kk << 2) + 2 + half];
                const int kb = ks * 64 + kk * 32;       // element offset in B rows
                FragAB bb;
                const u32x4* bp;
                bp = (const u32x4*)(b0 + kb); bb.q[0] = bp[half]; bb.q[1] = bp[2 + half];
                acc0.v = __builtin_amdgcn_wmma_f32_16x16x32_bf16(false, a.v, false, bb.v, (short)0, acc0.v, false, false);
                bp = (const u32x4*)(b1 + kb); bb.q[0] = bp[half]; bb.q[1] = bp[2 + half];
                acc1.v = __builtin_amdgcn_wmma_f32_16x16x32_bf16(false, a.v, false, bb.v, (short)0, acc1.v, false, false);
                bp = (const u32x4*)(b2 + kb); bb.q[0] = bp[half]; bb.q[1] = bp[2 + half];
                acc2.v = __builtin_amdgcn_wmma_f32_16x16x32_bf16(false, a.v, false, bb.v, (short)0, acc2.v, false, false);
                bp = (const u32x4*)(b3 + kb); bb.q[0] = bp[half]; bb.q[1] = bp[2 + half];
                acc3.v = __builtin_amdgcn_wmma_f32_16x16x32_bf16(false, a.v, false, bb.v, (short)0, acc3.v, false, false);
            }
        }
    }
    if (active) {
        // C/D layout: VGPR r -> M = r + 8*half, N = lane%16
        float* cb = C + (size_t)(m0 + (half << 3)) * ldc + grp * 64 + mlan;
#pragma unroll
        for (int r = 0; r < 8; ++r) {
            cb[(size_t)r * ldc +  0] = acc0.f[r];
            cb[(size_t)r * ldc + 16] = acc1.f[r];
            cb[(size_t)r * ldc + 32] = acc2.f[r];
            cb[(size_t)r * ldc + 48] = acc3.f[r];
        }
    }
}

// ---------------------------------------------------------------------------
// Weight packing: f32 [N,K] -> bf16 [Npad,Kpad] zero-padded
// ---------------------------------------------------------------------------
__global__ void pack_bf16_k(const float* __restrict__ src, unsigned short* __restrict__ dst,
                            int N, int K, int Npad, int Kpad)
{
    int idx = blockIdx.x * 256 + threadIdx.x;
    int total = Npad * Kpad;
    if (idx >= total) return;
    int k = idx % Kpad, n = idx / Kpad;
    float v = (n < N && k < K) ? src[(size_t)n * K + k] : 0.0f;
    dst[idx] = f32_to_bf16(v);
}

// ---------------------------------------------------------------------------
// CNN front-end
// ---------------------------------------------------------------------------
__global__ void flux_k(const float* __restrict__ x, float* __restrict__ out)
{
    int idx = blockIdx.x * 256 + threadIdx.x;           // B*160*T
    if (idx >= BB * 160 * TT) return;
    int t = idx % TT; int r = (idx / TT) % 160; int b = idx / (TT * 160);
    float v;
    if (r < 80) {
        v = x[((size_t)b * 80 + r) * TT + t];
    } else {
        int m = r - 80;
        float d = (t > 0) ? x[((size_t)b * 80 + m) * TT + t] - x[((size_t)b * 80 + m) * TT + t - 1]
                          : 0.0f;
        v = fmaxf(d, 0.0f);
    }
    out[idx] = v;
}

__global__ void conv3x3_k(const float* __restrict__ in, const float* __restrict__ w,
                          const float* __restrict__ bias, float* __restrict__ out,
                          int Cin, int H, int W, int doRelu, int total)
{
    int idx = blockIdx.x * 256 + threadIdx.x;
    if (idx >= total) return;
    int x0 = idx % W;
    int y0 = (idx / W) % H;
    int co = (idx / (W * H)) % CC;
    int b  = idx / (W * H * CC);
    float acc = bias[co];
    for (int ci = 0; ci < Cin; ++ci) {
        const float* ip = in + (size_t)(b * Cin + ci) * H * W;
        const float* wp = w + (size_t)(co * Cin + ci) * 9;
#pragma unroll
        for (int ky = 0; ky < 3; ++ky) {
            int yy = y0 + ky - 1;
            if ((unsigned)yy >= (unsigned)H) continue;
#pragma unroll
            for (int kx = 0; kx < 3; ++kx) {
                int xx = x0 + kx - 1;
                if ((unsigned)xx >= (unsigned)W) continue;
                acc += wp[ky * 3 + kx] * ip[(size_t)yy * W + xx];
            }
        }
    }
    out[idx] = doRelu ? fmaxf(acc, 0.0f) : acc;
}

__global__ void skipadd_k(const float* __restrict__ r2, const float* __restrict__ h0,
                          const float* __restrict__ ws, const float* __restrict__ bs,
                          float* __restrict__ out)
{
    int idx = blockIdx.x * 256 + threadIdx.x;           // B*C*160*T
    if (idx >= BB * CC * 160 * TT) return;
    int hw = idx % (160 * TT);
    int c  = (idx / (160 * TT)) % CC;
    int b  = idx / (160 * TT * CC);
    float skip = h0[(size_t)b * 160 * TT + hw] * ws[c] + bs[c];
    out[idx] = fmaxf(r2[idx] + skip, 0.0f);
}

__global__ void addrelu_k(const float* __restrict__ a, const float* __restrict__ b,
                          float* __restrict__ out, int total)
{
    int idx = blockIdx.x * 256 + threadIdx.x;
    if (idx >= total) return;
    out[idx] = fmaxf(a[idx] + b[idx], 0.0f);
}

__global__ void maxpool_k(const float* __restrict__ in, float* __restrict__ out,
                          int Hin, int W, int total)
{
    int idx = blockIdx.x * 256 + threadIdx.x;           // B*C*(Hin/2)*W
    if (idx >= total) return;
    int Ho = Hin >> 1;
    int t  = idx % W;
    int f  = (idx / W) % Ho;
    int bc = idx / (W * Ho);
    const float* p = in + ((size_t)bc * Hin + 2 * f) * W + t;
    out[idx] = fmaxf(p[0], p[W]);
}

__global__ void reshape_k(const float* __restrict__ p2, float* __restrict__ seqf,
                          unsigned short* __restrict__ seqb)
{
    int idx = blockIdx.x * 256 + threadIdx.x;           // B*24*40*T
    if (idx >= BB * CC * 40 * TT) return;
    int t = idx % TT;
    int f = (idx / TT) % 40;
    int c = (idx / (TT * 40)) % CC;
    int b = idx / (TT * 40 * CC);
    float v = p2[idx];
    size_t o = (size_t)(b * TT + t) * DM + c * 40 + f;
    seqf[o] = v;
    seqb[o] = f32_to_bf16(v);
}

// ---------------------------------------------------------------------------
// Mamba pieces
// ---------------------------------------------------------------------------
__global__ void dwconv_silu_k(const float* __restrict__ xz, const float* __restrict__ cw,
                              const float* __restrict__ cb, float* __restrict__ xc,
                              unsigned short* __restrict__ xcb)
{
    int idx = blockIdx.x * 256 + threadIdx.x;           // M * DI
    if (idx >= MM * DI) return;
    int d  = idx % DI;
    int bt = idx / DI;
    int b = bt >> 9, t = bt & 511;
    float acc = cb[d];
#pragma unroll
    for (int j = 0; j < 4; ++j) {
        int tt = t - 3 + j;
        if (tt >= 0) acc += cw[d * 4 + j] * xz[(size_t)(b * TT + tt) * 3840 + d];
    }
    float s = acc * sigm(acc);                          // SiLU
    xc[idx]  = s;
    xcb[idx] = f32_to_bf16(s);
}

__global__ void dtin_k(const float* __restrict__ dbc, unsigned short* __restrict__ dtin)
{
    int idx = blockIdx.x * 256 + threadIdx.x;           // M * 64
    if (idx >= MM * 64) return;
    int r  = idx & 63;
    int bt = idx >> 6;
    float v = (r < 60) ? dbc[(size_t)bt * 128 + r] : 0.0f;
    dtin[idx] = f32_to_bf16(v);
}

__global__ void dtsp_k(float* __restrict__ dt, const float* __restrict__ dtb)
{
    int idx = blockIdx.x * 256 + threadIdx.x;           // M * DI
    if (idx >= MM * DI) return;
    int d = idx % DI;
    float v = dt[idx] + dtb[d];
    dt[idx] = (v > 20.0f) ? v : log1pf(expf(v));        // softplus
}

__global__ void scan_k(const float* __restrict__ xc, const float* __restrict__ dt,
                       const float* __restrict__ dbc, const float* __restrict__ A_log,
                       const float* __restrict__ Dp, float* __restrict__ ys)
{
    int idx = blockIdx.x * 256 + threadIdx.x;           // B * DI independent scans
    if (idx >= BB * DI) return;
    int d = idx % DI;
    int b = idx / DI;
    float Aarr[16], h[16];
#pragma unroll
    for (int n = 0; n < 16; ++n) { Aarr[n] = -expf(A_log[(size_t)d * 16 + n]); h[n] = 0.0f; }
    float Dv = Dp[d];
    for (int t = 0; t < TT; ++t) {
        size_t bt = (size_t)b * TT + t;
        float dtv = dt[bt * DI + d];
        float u   = xc[bt * DI + d];
        const float* Bt = dbc + bt * 128 + 60;
        const float* Ct = dbc + bt * 128 + 76;
        float y = 0.0f;
        float du = dtv * u;
#pragma unroll
        for (int n = 0; n < 16; ++n) {
            float dA = expf(dtv * Aarr[n]);
            h[n] = fmaf(dA, h[n], du * Bt[n]);
            y    = fmaf(h[n], Ct[n], y);
        }
        ys[bt * DI + d] = fmaf(u, Dv, y);
    }
}

__global__ void gate_k(const float* __restrict__ ys, const float* __restrict__ xz,
                       unsigned short* __restrict__ yb)
{
    int idx = blockIdx.x * 256 + threadIdx.x;           // M * DI
    if (idx >= MM * DI) return;
    int d  = idx % DI;
    int bt = idx / DI;
    float z = xz[(size_t)bt * 3840 + DI + d];
    float v = ys[idx] * (z * sigm(z));                  // y * silu(z)
    yb[idx] = f32_to_bf16(v);
}

__global__ __launch_bounds__(256) void rmsnorm_k(const float* __restrict__ o,
                                                 float* __restrict__ seqf,
                                                 unsigned short* __restrict__ seqb,
                                                 const float* __restrict__ w)
{
    __shared__ float red[256];
    int row = blockIdx.x, tid = threadIdx.x;
    const float* orow = o + (size_t)row * DM;
    float* srow = seqf + (size_t)row * DM;
    float loc = 0.0f;
    for (int d = tid; d < DM; d += 256) { float v = orow[d] + srow[d]; loc += v * v; }
    red[tid] = loc;
    __syncthreads();
    for (int s = 128; s > 0; s >>= 1) {
        if (tid < s) red[tid] += red[tid + s];
        __syncthreads();
    }
    float scale = rsqrtf(red[0] / (float)DM + 1e-5f);
    for (int d = tid; d < DM; d += 256) {
        float v = (orow[d] + srow[d]) * scale * w[d];
        srow[d] = v;
        seqb[(size_t)row * DM + d] = f32_to_bf16(v);
    }
}

__global__ void fc_out_k(const float* __restrict__ logits, const float* __restrict__ fcb,
                         float* __restrict__ out)
{
    int idx = blockIdx.x * 256 + threadIdx.x;           // B * 10 * T
    if (idx >= BB * 10 * TT) return;
    int t = idx % TT;
    int c = (idx / TT) % 10;
    int b = idx / (TT * 10);
    float v = logits[(size_t)(b * TT + t) * 64 + c] + fcb[c];
    out[idx] = fmaxf(v, 0.0f);
}

// ---------------------------------------------------------------------------
// Host launcher
// ---------------------------------------------------------------------------
extern "C" void kernel_launch(void* const* d_in, const int* in_sizes, int n_in,
                              void* d_out, int out_size, void* d_ws, size_t ws_size,
                              hipStream_t stream)
{
    (void)in_sizes; (void)n_in; (void)out_size;
    if (ws_size < O_END) return;

    const float* x       = (const float*)d_in[0];
    const float* cb1_w1  = (const float*)d_in[1];
    const float* cb1_b1  = (const float*)d_in[2];
    const float* cb1_w2  = (const float*)d_in[3];
    const float* cb1_b2  = (const float*)d_in[4];
    const float* cb1_ws  = (const float*)d_in[5];
    const float* cb1_bs  = (const float*)d_in[6];
    const float* cb2_w1  = (const float*)d_in[7];
    const float* cb2_b1  = (const float*)d_in[8];
    const float* cb2_w2  = (const float*)d_in[9];
    const float* cb2_b2  = (const float*)d_in[10];
    const float* in_w    = (const float*)d_in[11];
    const float* conv_w  = (const float*)d_in[12];
    const float* conv_b  = (const float*)d_in[13];
    const float* xp_w    = (const float*)d_in[14];
    const float* dtp_w   = (const float*)d_in[15];
    const float* dtp_b   = (const float*)d_in[16];
    const float* A_log   = (const float*)d_in[17];
    const float* D_par   = (const float*)d_in[18];
    const float* out_w   = (const float*)d_in[19];
    const float* norm_w  = (const float*)d_in[20];
    const float* fc_w    = (const float*)d_in[21];
    const float* fc_b    = (const float*)d_in[22];

    char* ws = (char*)d_ws;
    unsigned short* WBIN  = (unsigned short*)(ws + O_WB_IN);
    unsigned short* WBX   = (unsigned short*)(ws + O_WB_X);
    unsigned short* WBDT  = (unsigned short*)(ws + O_WB_DT);
    unsigned short* WBOUT = (unsigned short*)(ws + O_WB_OUT);
    unsigned short* WBFC  = (unsigned short*)(ws + O_WB_FC);
    float*          SEQF  = (float*)(ws + O_SEQF);
    unsigned short* SEQB  = (unsigned short*)(ws + O_SEQB);
    float* XZ   = (float*)(ws + O_XZ);
    float* XC   = (float*)(ws + O_XC);
    unsigned short* XCB = (unsigned short*)(ws + O_XCB);
    float* DBC  = (float*)(ws + O_DBC);
    unsigned short* DTI = (unsigned short*)(ws + O_DTI);
    float* DTP  = (float*)(ws + O_DTP);
    float* YS   = (float*)(ws + O_YS);
    unsigned short* YB = (unsigned short*)(ws + O_YB);
    float* OP   = (float*)(ws + O_OP);
    float* LOG  = (float*)(ws + O_LOG);
    float* HFLUX = (float*)(ws + O_HFLUX);
    float* R1 = (float*)(ws + O_R1);
    float* R2 = (float*)(ws + O_R2);
    float* R3 = (float*)(ws + O_R3);
    float* R4 = (float*)(ws + O_R4);
    float* H2 = (float*)(ws + O_H2);
    float* P2 = (float*)(ws + O_P2);

    dim3 blk(256);
#define G1(n) dim3((unsigned)(((n) + 255) / 256))

    // ---- pack all GEMM weights to padded bf16 panels (N padded to 64) ----
    for (int l = 0; l < 3; ++l) {
        pack_bf16_k<<<G1(3840 * 960), blk, 0, stream>>>(in_w  + (size_t)l * 3840 * 960,
                                                        WBIN  + (size_t)l * 3840 * 960, 3840, 960, 3840, 960);
        pack_bf16_k<<<G1(128 * 1920), blk, 0, stream>>>(xp_w  + (size_t)l * 92 * 1920,
                                                        WBX   + (size_t)l * 128 * 1920, 92, 1920, 128, 1920);
        pack_bf16_k<<<G1(1920 * 64),  blk, 0, stream>>>(dtp_w + (size_t)l * 1920 * 60,
                                                        WBDT  + (size_t)l * 1920 * 64, 1920, 60, 1920, 64);
        pack_bf16_k<<<G1(960 * 1920), blk, 0, stream>>>(out_w + (size_t)l * 960 * 1920,
                                                        WBOUT + (size_t)l * 960 * 1920, 960, 1920, 960, 1920);
    }
    pack_bf16_k<<<G1(64 * 960), blk, 0, stream>>>(fc_w, WBFC, 10, 960, 64, 960);

    // ---- CNN front-end ----
    flux_k<<<G1(BB * 160 * TT), blk, 0, stream>>>(x, HFLUX);
    int n1 = BB * CC * 160 * TT;                      // 7,864,320
    conv3x3_k<<<G1(n1), blk, 0, stream>>>(HFLUX, cb1_w1, cb1_b1, R1, 1, 160, TT, 1, n1);
    conv3x3_k<<<G1(n1), blk, 0, stream>>>(R1, cb1_w2, cb1_b2, R2, 24, 160, TT, 0, n1);
    skipadd_k<<<G1(n1), blk, 0, stream>>>(R2, HFLUX, cb1_ws, cb1_bs, R1);         // H1 -> R1
    int n2 = BB * CC * 80 * TT;
    maxpool_k<<<G1(n2), blk, 0, stream>>>(R1, R2, 160, TT, n2);                   // P1 -> R2
    conv3x3_k<<<G1(n2), blk, 0, stream>>>(R2, cb2_w1, cb2_b1, R3, 24, 80, TT, 1, n2);
    conv3x3_k<<<G1(n2), blk, 0, stream>>>(R3, cb2_w2, cb2_b2, R4, 24, 80, TT, 0, n2);
    addrelu_k<<<G1(n2), blk, 0, stream>>>(R4, R2, H2, n2);
    int n3 = BB * CC * 40 * TT;
    maxpool_k<<<G1(n3), blk, 0, stream>>>(H2, P2, 80, TT, n3);
    reshape_k<<<G1(n3), blk, 0, stream>>>(P2, SEQF, SEQB);

    // ---- Mamba layers ----
    for (int l = 0; l < 3; ++l) {
        // xz = seq @ in_proj^T : [2048,3840]  (nTiles=240 -> 60 groups)
        gemm_bf16_wmma<<<dim3(MM / 16, 8), blk, 0, stream>>>(
            SEQB, WBIN + (size_t)l * 3840 * 960, XZ, 960, 960, 3840, 30, 240);
        // causal depthwise conv1d + SiLU
        dwconv_silu_k<<<G1(MM * DI), blk, 0, stream>>>(
            XZ, conv_w + (size_t)l * DI * 4, conv_b + (size_t)l * DI, XC, XCB);
        // dbc = xc @ x_proj^T : [2048,128]  (N padded 92->128)
        gemm_bf16_wmma<<<dim3(MM / 16, 1), blk, 0, stream>>>(
            XCB, WBX + (size_t)l * 128 * 1920, DBC, 1920, 1920, 128, 60, 8);
        // dt input (60 -> padded 64, bf16)
        dtin_k<<<G1(MM * 64), blk, 0, stream>>>(DBC, DTI);
        // dt_pre = dt_in @ dt_proj^T : [2048,1920]  (nTiles=120 -> 30 groups)
        gemm_bf16_wmma<<<dim3(MM / 16, 4), blk, 0, stream>>>(
            DTI, WBDT + (size_t)l * 1920 * 64, DTP, 64, 64, 1920, 2, 120);
        // dt = softplus(dt_pre + dtb), in place
        dtsp_k<<<G1(MM * DI), blk, 0, stream>>>(DTP, dtp_b + (size_t)l * DI);
        // selective scan
        scan_k<<<G1(BB * DI), blk, 0, stream>>>(
            XC, DTP, DBC, A_log + (size_t)l * DI * 16, D_par + (size_t)l * DI, YS);
        // y * silu(z) -> bf16
        gate_k<<<G1(MM * DI), blk, 0, stream>>>(YS, XZ, YB);
        // o = y @ out_proj^T : [2048,960]  (nTiles=60 -> 15 groups)
        gemm_bf16_wmma<<<dim3(MM / 16, 2), blk, 0, stream>>>(
            YB, WBOUT + (size_t)l * 960 * 1920, OP, 1920, 1920, 960, 60, 60);
        // seq = rmsnorm(o + seq)
        rmsnorm_k<<<dim3(MM), blk, 0, stream>>>(OP, SEQF, SEQB, norm_w + (size_t)l * DM);
    }

    // ---- classifier head ----
    gemm_bf16_wmma<<<dim3(MM / 16, 1), blk, 0, stream>>>(SEQB, WBFC, LOG, 960, 960, 64, 30, 4);
    fc_out_k<<<G1(BB * 10 * TT), blk, 0, stream>>>(LOG, fc_b, (float*)d_out);
#undef G1
}
